// Mymodel_90666759618894
// MI455X (gfx1250) — matrix operand order repaired and checked by reading, not compiled, not run on hardware
//
#include <hip/hip_runtime.h>

#define S_ 128
#define T_ 64
#define B_ 32
#define E_ 256
#define H_ 512
#define V_ 30000

typedef __attribute__((ext_vector_type(16))) __bf16 v16bf;
typedef __attribute__((ext_vector_type(8)))  float  v8f;

// ---------------------------------------------------------------- helpers ---
__device__ __forceinline__ __bf16 f2bf(float f) {
  union { float f; unsigned u; } x; x.f = f;
  unsigned r = x.u + 0x7FFFu + ((x.u >> 16) & 1u);   // round-to-nearest-even
  unsigned short hb = (unsigned short)(r >> 16);
  return __builtin_bit_cast(__bf16, hb);
}
__device__ __forceinline__ float sig_(float x) { return 1.0f / (1.0f + __expf(-x)); }

__device__ __forceinline__ v8f wmma_bf16(v16bf a, v16bf b, v8f c) {
  return __builtin_amdgcn_wmma_f32_16x16x32_bf16(false, a, false, b, (short)0, c,
                                                 false, false);
}

// A fragment: 16x32 fp32 tile (rows m0..m0+15, K k0..k0+31) -> bf16 in-reg.
// A layout (matches B): lane hi*16+m : element e -> K=hi*16+e
__device__ __forceinline__ v16bf load_afrag(const float* __restrict__ A, int lda,
                                            int m0, int k0) {
  const int lane = threadIdx.x & 31;
  const int hi = lane >> 4, m = lane & 15;
  const float* row = A + (size_t)(m0 + m) * lda + k0 + hi * 16;
  float4 p0 = *(const float4*)(row);
  float4 p1 = *(const float4*)(row + 4);
  float4 p2 = *(const float4*)(row + 8);
  float4 p3 = *(const float4*)(row + 12);
  v16bf a;
  a[0]=f2bf(p0.x);  a[1]=f2bf(p0.y);  a[2]=f2bf(p0.z);  a[3]=f2bf(p0.w);
  a[4]=f2bf(p1.x);  a[5]=f2bf(p1.y);  a[6]=f2bf(p1.z);  a[7]=f2bf(p1.w);
  a[8]=f2bf(p2.x);  a[9]=f2bf(p2.y);  a[10]=f2bf(p2.z); a[11]=f2bf(p2.w);
  a[12]=f2bf(p3.x); a[13]=f2bf(p3.y); a[14]=f2bf(p3.z); a[15]=f2bf(p3.w);
  return a;
}

// B fragment: 32x16 from bf16 weights W[N,K] row-major (computing A @ W^T).
// lane hi*16+n holds column n, K = k0+hi*16 .. +15 (16 contiguous bf16 = 32B).
__device__ __forceinline__ v16bf load_bfrag(const __bf16* __restrict__ W, int ldk,
                                            int n0, int k0) {
  const int lane = threadIdx.x & 31;
  const int hi = lane >> 4, n = lane & 15;
  return *(const v16bf*)(W + (size_t)(n0 + n) * ldk + k0 + hi * 16);
}

// ------------------------------------------------------------ WMMA GEMM -----
// C[M,N] = A[M,K](f32) @ W[N,K](bf16)^T (+ bias[N]).
// One wave = 32x32 macro-tile (2 A-frags x 2 B-frags -> 4 WMMA / K-step).
// grid = (ceil(N/32), M/32), block = 32.  N must be a multiple of 16.
__global__ void wmma_gemm_nt(const float* __restrict__ A, int lda,
                             const __bf16* __restrict__ W, int ldk,
                             float* __restrict__ C, int ldc,
                             const float* __restrict__ bias, int K, int N) {
  const int n0 = blockIdx.x * 32;
  const int m0 = blockIdx.y * 32;
  const int lane = threadIdx.x & 31;
  const bool hasN1 = (n0 + 16) < N;        // block-uniform edge guard (N=30000)
  v8f acc00 = {}, acc01 = {}, acc10 = {}, acc11 = {};
  for (int k0 = 0; k0 < K; k0 += 32) {
    if (k0 + 32 < K) {
      __builtin_prefetch(A + (size_t)(m0 + (lane & 15)) * lda + k0 + 32, 0, 3);
      __builtin_prefetch(W + (size_t)(n0 + (lane & 15)) * ldk + k0 + 32, 0, 3);
    }
    v16bf a0 = load_afrag(A, lda, m0, k0);
    v16bf a1 = load_afrag(A, lda, m0 + 16, k0);
    v16bf b0 = load_bfrag(W, ldk, n0, k0);
    acc00 = wmma_bf16(a0, b0, acc00);
    acc10 = wmma_bf16(a1, b0, acc10);
    if (hasN1) {
      v16bf b1 = load_bfrag(W, ldk, n0 + 16, k0);
      acc01 = wmma_bf16(a0, b1, acc01);
      acc11 = wmma_bf16(a1, b1, acc11);
    }
  }
  const int hi = lane >> 4, n = lane & 15;
  const float bv0 = bias ? bias[n0 + n] : 0.0f;
  const float bv1 = (bias && hasN1) ? bias[n0 + 16 + n] : 0.0f;
#pragma unroll
  for (int r = 0; r < 8; ++r) {
    const int r0 = m0 + hi * 8 + r, r1 = m0 + 16 + hi * 8 + r;
    C[(size_t)r0 * ldc + n0 + n] = acc00[r] + bv0;
    C[(size_t)r1 * ldc + n0 + n] = acc10[r] + bv0;
    if (hasN1) {
      C[(size_t)r0 * ldc + n0 + 16 + n] = acc01[r] + bv1;
      C[(size_t)r1 * ldc + n0 + 16 + n] = acc11[r] + bv1;
    }
  }
}

// ------------------------------------------------------ fused LSTM step -----
// z = xpre[t] + h_in @ Whh^T ; gates i,f,g,o ; writes h_out, c_out (+hs copy).
// One wave owns ALL 32 batch rows x 16 hidden units: 2 A-frags shared across
// the 4 gate B-frags -> 8 WMMA / K-step.  grid = (H/16), block = 32.
__global__ void lstm_step(const float* __restrict__ h_in,
                          const float* __restrict__ c_in,
                          const __bf16* __restrict__ Whh,   // [4H,H] bf16
                          const float* __restrict__ xpre,   // [B,4H]
                          float* __restrict__ h_out,
                          float* __restrict__ c_out,
                          float* __restrict__ hs_out, int hs_ld, int hs_off,
                          int H) {
  const int n0 = blockIdx.x * 16;
  const int lane = threadIdx.x & 31;
  v8f ai0 = {}, af0 = {}, ag0 = {}, ao0 = {};
  v8f ai1 = {}, af1 = {}, ag1 = {}, ao1 = {};
  for (int k0 = 0; k0 < H; k0 += 32) {
    v16bf a0 = load_afrag(h_in, H, 0, k0);
    v16bf a1 = load_afrag(h_in, H, 16, k0);
    v16bf bi = load_bfrag(Whh, H, n0,         k0);
    v16bf bf = load_bfrag(Whh, H, n0 + H,     k0);
    v16bf bg = load_bfrag(Whh, H, n0 + 2 * H, k0);
    v16bf bo = load_bfrag(Whh, H, n0 + 3 * H, k0);
    ai0 = wmma_bf16(a0, bi, ai0);  ai1 = wmma_bf16(a1, bi, ai1);
    af0 = wmma_bf16(a0, bf, af0);  af1 = wmma_bf16(a1, bf, af1);
    ag0 = wmma_bf16(a0, bg, ag0);  ag1 = wmma_bf16(a1, bg, ag1);
    ao0 = wmma_bf16(a0, bo, ao0);  ao1 = wmma_bf16(a1, bo, ao1);
  }
  const int hi = lane >> 4, n = lane & 15;
  const int col = n0 + n;
#pragma unroll
  for (int mb = 0; mb < 2; ++mb) {
    const v8f& zi8 = mb ? ai1 : ai0;
    const v8f& zf8 = mb ? af1 : af0;
    const v8f& zg8 = mb ? ag1 : ag0;
    const v8f& zo8 = mb ? ao1 : ao0;
#pragma unroll
    for (int r = 0; r < 8; ++r) {
      const int row = mb * 16 + hi * 8 + r;
      const float* xp = xpre + (size_t)row * 4 * H;
      float zi = zi8[r] + xp[col];
      float zf = zf8[r] + xp[H + col];
      float zg = zg8[r] + xp[2 * H + col];
      float zo = zo8[r] + xp[3 * H + col];
      float c  = sig_(zf) * c_in[(size_t)row * H + col] + sig_(zi) * tanhf(zg);
      float h  = sig_(zo) * tanhf(c);
      c_out[(size_t)row * H + col] = c;
      h_out[(size_t)row * H + col] = h;
      if (hs_out) hs_out[(size_t)row * hs_ld + hs_off + col] = h;
    }
  }
}

// ------------------------------------------------- Bahdanau attention -------
// e[s] = u_a . tanh(WaH[s,b,:] + vh[b,:]); alpha = softmax(e * mask, over S);
// ct[b,:] = sum_s alpha[s] * hs[s,b,:].   grid = B, block = 128 (=S_).
__global__ void attn_kernel(const float* __restrict__ WaH,   // [S,B,H]
                            const float* __restrict__ vh,    // [B,H]
                            const float* __restrict__ u_a,   // [H]
                            const float* __restrict__ s_mask,// [S,B]
                            const float* __restrict__ hs,    // [S,B,2H]
                            float* __restrict__ ct, int ld_ct) {
  __shared__ float e[S_];
  __shared__ float red[S_];
  const int b = blockIdx.x;
  const int tid = threadIdx.x;
  {
    const float* wa = WaH + ((size_t)tid * B_ + b) * H_;
    const float* vhb = vh + (size_t)b * H_;
    float acc = 0.f;
    for (int h = 0; h < H_; ++h) acc += u_a[h] * tanhf(wa[h] + vhb[h]);
    e[tid] = acc * s_mask[tid * B_ + b];   // reference multiplies logits by mask
  }
  __syncthreads();
  red[tid] = e[tid];
  __syncthreads();
  for (int off = S_ / 2; off > 0; off >>= 1) {
    if (tid < off) red[tid] = fmaxf(red[tid], red[tid + off]);
    __syncthreads();
  }
  const float mx = red[0];
  __syncthreads();
  const float ex = __expf(e[tid] - mx);
  red[tid] = ex;
  __syncthreads();
  for (int off = S_ / 2; off > 0; off >>= 1) {
    if (tid < off) red[tid] += red[tid + off];
    __syncthreads();
  }
  const float inv = 1.0f / red[0];
  __syncthreads();
  e[tid] = ex * inv;                        // alpha[s]
  __syncthreads();
  for (int h2 = tid; h2 < 2 * H_; h2 += S_) {
    float s = 0.f;
    for (int sx = 0; sx < S_; ++sx)
      s += e[sx] * hs[((size_t)sx * B_ + b) * (2 * H_) + h2];
    ct[(size_t)b * ld_ct + h2] = s;
  }
}

// ------------------------------------------------------- softmax over V -----
__global__ void softmax_rows(const float* __restrict__ logits,
                             float* __restrict__ p, int V) {
  __shared__ float red[256];
  const int b = blockIdx.x, tid = threadIdx.x;
  const float* row = logits + (size_t)b * V;
  float mx = -3.4e38f;
  for (int i = tid; i < V; i += 256) mx = fmaxf(mx, row[i]);
  red[tid] = mx; __syncthreads();
  for (int off = 128; off > 0; off >>= 1) {
    if (tid < off) red[tid] = fmaxf(red[tid], red[tid + off]);
    __syncthreads();
  }
  mx = red[0]; __syncthreads();
  float sum = 0.f;
  for (int i = tid; i < V; i += 256) sum += __expf(row[i] - mx);
  red[tid] = sum; __syncthreads();
  for (int off = 128; off > 0; off >>= 1) {
    if (tid < off) red[tid] += red[tid + off];
    __syncthreads();
  }
  const float inv = 1.0f / red[0];
  float* prow = p + (size_t)b * V;
  for (int i = tid; i < V; i += 256) prow[i] = __expf(row[i] - mx) * inv;
}

// ------------------------------------------------------------- loss ---------
__global__ void loss_kernel(const float* __restrict__ p,      // [T,B,V]
                            const int* __restrict__ tind,     // [T,B]
                            const float* __restrict__ tmask,  // [T,B]
                            float* __restrict__ loss) {
  const int b = threadIdx.x;                 // 32 threads
  float num = 0.f, den = 0.f;
  for (int t = 0; t < T_; ++t) {
    const int tn = (t < T_ - 1) ? t + 1 : T_ - 1;
    const int y = tind[tn * B_ + b];
    const float pv = p[((size_t)t * B_ + b) * V_ + y];
    const float m = tmask[t * B_ + b];
    num += -__logf(pv) * m;
    den += m;
  }
  loss[b] = num / den;
}

// ------------------------------------------------------- small utilities ----
__global__ void pack_bf16(const float* __restrict__ src, __bf16* __restrict__ dst,
                          size_t n) {
  size_t i = (size_t)blockIdx.x * 256 + threadIdx.x;
  if (i < n) dst[i] = f2bf(src[i]);
}
__global__ void pack_bf16_cols(const float* __restrict__ src, int sld,
                               __bf16* __restrict__ dst, int dld, int coloff,
                               int rows, int cols) {
  size_t i = (size_t)blockIdx.x * 256 + threadIdx.x;
  if (i >= (size_t)rows * cols) return;
  int r = (int)(i / cols), c = (int)(i % cols);
  dst[(size_t)r * dld + coloff + c] = f2bf(src[(size_t)r * sld + c]);
}
__global__ void copy_cols_f32(const float* __restrict__ src, int sld,
                              float* __restrict__ dst, int dld, int coloff,
                              int rows, int cols) {
  int i = blockIdx.x * 256 + threadIdx.x;
  if (i >= rows * cols) return;
  int r = i / cols, c = i % cols;
  dst[(size_t)r * dld + coloff + c] = src[(size_t)r * sld + c];
}
__global__ void tanh_vec(const float* __restrict__ src, float* __restrict__ dst,
                         int n) {
  int i = blockIdx.x * 256 + threadIdx.x;
  if (i < n) dst[i] = tanhf(src[i]);
}
__global__ void zero_vec(float* __restrict__ p, int n) {
  int i = blockIdx.x * 256 + threadIdx.x;
  if (i < n) p[i] = 0.f;
}

// ---------------------------------------------------------------- host ------
extern "C" void kernel_launch(void* const* d_in, const int* in_sizes, int n_in,
                              void* d_out, int out_size, void* d_ws, size_t ws_size,
                              hipStream_t stream) {
  (void)in_sizes; (void)n_in; (void)out_size; (void)ws_size;
  const float* s_emb  = (const float*)d_in[0];
  const float* t_emb  = (const float*)d_in[1];
  const float* s_mask = (const float*)d_in[2];
  const float* t_mask = (const float*)d_in[3];
  const int*   t_ind  = (const int*)  d_in[4];
  const float* Wih_f  = (const float*)d_in[5];
  const float* Whh_f  = (const float*)d_in[6];
  const float* b_f    = (const float*)d_in[7];
  const float* Wih_b  = (const float*)d_in[8];
  const float* Whh_b  = (const float*)d_in[9];
  const float* b_b    = (const float*)d_in[10];
  const float* W_init = (const float*)d_in[11];
  const float* b_init = (const float*)d_in[12];
  const float* Wih_d  = (const float*)d_in[13];
  const float* Whh_d  = (const float*)d_in[14];
  const float* b_d    = (const float*)d_in[15];
  const float* w_a    = (const float*)d_in[16];
  const float* v_a    = (const float*)d_in[17];
  const float* u_a    = (const float*)d_in[18];
  const float* w_b    = (const float*)d_in[19];
  const float* v_b    = (const float*)d_in[20];

  float* out  = (float*)d_out;
  float* loss = out;          // [B]
  float* p    = out + B_;     // [T,B,V]

  char* wp = (char*)d_ws;
  auto carve = [&](size_t bytes) -> char* {
    char* r = wp; wp += (bytes + 255) & ~(size_t)255; return r;
  };
  float*  Xf    = (float*) carve((size_t)S_ * B_ * 4 * H_ * 4);
  float*  Xb    = (float*) carve((size_t)S_ * B_ * 4 * H_ * 4);
  float*  Xd    = (float*) carve((size_t)T_ * B_ * 4 * H_ * 4);
  float*  hsb   = (float*) carve((size_t)S_ * B_ * 2 * H_ * 4);
  float*  WaH   = (float*) carve((size_t)S_ * B_ * H_ * 4);
  float*  hpf   = (float*) carve((size_t)2 * B_ * H_ * 4);
  float*  cpf   = (float*) carve((size_t)2 * B_ * H_ * 4);
  float*  hpb   = (float*) carve((size_t)2 * B_ * H_ * 4);
  float*  cpb   = (float*) carve((size_t)2 * B_ * H_ * 4);
  float*  hdec  = (float*) carve((size_t)2 * B_ * H_ * 4);
  float*  cdec  = (float*) carve((size_t)2 * B_ * H_ * 4);
  float*  Ccat  = (float*) carve((size_t)B_ * 2 * H_ * 4);
  float*  vh    = (float*) carve((size_t)B_ * H_ * 4);
  float*  Acat  = (float*) carve((size_t)B_ * (3 * H_) * 4);     // [h | ct] 1536
  float*  lgts  = (float*) carve((size_t)B_ * V_ * 4);
  __bf16* bWihF = (__bf16*)carve((size_t)4 * H_ * E_ * 2);
  __bf16* bWihB = (__bf16*)carve((size_t)4 * H_ * E_ * 2);
  __bf16* bWihD = (__bf16*)carve((size_t)4 * H_ * E_ * 2);
  __bf16* bWhhF = (__bf16*)carve((size_t)4 * H_ * H_ * 2);
  __bf16* bWhhB = (__bf16*)carve((size_t)4 * H_ * H_ * 2);
  __bf16* bWhhD = (__bf16*)carve((size_t)4 * H_ * H_ * 2);
  __bf16* bWini = (__bf16*)carve((size_t)H_ * 2 * H_ * 2);
  __bf16* bWa   = (__bf16*)carve((size_t)H_ * 2 * H_ * 2);
  __bf16* bVa   = (__bf16*)carve((size_t)H_ * H_ * 2);
  __bf16* bWcat = (__bf16*)carve((size_t)V_ * (3 * H_) * 2);     // [w_b | v_b]

  auto P = [&](const float* s, __bf16* d, size_t n) {
    pack_bf16<<<dim3((unsigned)((n + 255) / 256)), 256, 0, stream>>>(s, d, n);
  };
  // ---- one-time bf16 weight packing (amortized over all 64+256 GEMM steps)
  P(Wih_f, bWihF, (size_t)4 * H_ * E_);  P(Wih_b, bWihB, (size_t)4 * H_ * E_);
  P(Wih_d, bWihD, (size_t)4 * H_ * E_);
  P(Whh_f, bWhhF, (size_t)4 * H_ * H_);  P(Whh_b, bWhhB, (size_t)4 * H_ * H_);
  P(Whh_d, bWhhD, (size_t)4 * H_ * H_);
  P(W_init, bWini, (size_t)H_ * 2 * H_);
  P(w_a, bWa, (size_t)H_ * 2 * H_);      P(v_a, bVa, (size_t)H_ * H_);
  {
    size_t n1 = (size_t)V_ * H_, n2 = (size_t)V_ * 2 * H_;
    pack_bf16_cols<<<dim3((unsigned)((n1 + 255) / 256)), 256, 0, stream>>>(
        w_b, H_, bWcat, 3 * H_, 0, V_, H_);
    pack_bf16_cols<<<dim3((unsigned)((n2 + 255) / 256)), 256, 0, stream>>>(
        v_b, 2 * H_, bWcat, 3 * H_, H_, V_, 2 * H_);
  }

  // ---- hoisted input projections: X = emb @ Wih^T + b   (big WMMA GEMMs)
  wmma_gemm_nt<<<dim3(4 * H_ / 32, S_ * B_ / 32), 32, 0, stream>>>(
      s_emb, E_, bWihF, E_, Xf, 4 * H_, b_f, E_, 4 * H_);
  wmma_gemm_nt<<<dim3(4 * H_ / 32, S_ * B_ / 32), 32, 0, stream>>>(
      s_emb, E_, bWihB, E_, Xb, 4 * H_, b_b, E_, 4 * H_);
  wmma_gemm_nt<<<dim3(4 * H_ / 32, T_ * B_ / 32), 32, 0, stream>>>(
      t_emb, E_, bWihD, E_, Xd, 4 * H_, b_d, E_, 4 * H_);

  const int nz = B_ * H_;
  zero_vec<<<dim3((nz + 255) / 256), 256, 0, stream>>>(hpf, nz);
  zero_vec<<<dim3((nz + 255) / 256), 256, 0, stream>>>(cpf, nz);
  zero_vec<<<dim3((nz + 255) / 256), 256, 0, stream>>>(hpb, nz);
  zero_vec<<<dim3((nz + 255) / 256), 256, 0, stream>>>(cpb, nz);

  const dim3 lgrid(H_ / 16);
  // ---- forward encoder scan
  int cur = 0;
  for (int t = 0; t < S_; ++t) {
    lstm_step<<<lgrid, 32, 0, stream>>>(
        hpf + cur * nz, cpf + cur * nz, bWhhF, Xf + (size_t)t * B_ * 4 * H_,
        hpf + (1 - cur) * nz, cpf + (1 - cur) * nz,
        hsb + (size_t)t * B_ * 2 * H_, 2 * H_, 0, H_);
    cur = 1 - cur;
  }
  const float* c_fw_final = cpf + cur * nz;
  // ---- backward encoder scan
  cur = 0;
  for (int t = S_ - 1; t >= 0; --t) {
    lstm_step<<<lgrid, 32, 0, stream>>>(
        hpb + cur * nz, cpb + cur * nz, bWhhB, Xb + (size_t)t * B_ * 4 * H_,
        hpb + (1 - cur) * nz, cpb + (1 - cur) * nz,
        hsb + (size_t)t * B_ * 2 * H_, 2 * H_, H_, H_);
    cur = 1 - cur;
  }
  const float* c_bw_final = cpb + cur * nz;

  // ---- decoder init: cell0 = [c_fw|c_bw] @ W_init^T + b_init ; h0 = tanh(cell0)
  {
    int n = B_ * H_;
    copy_cols_f32<<<dim3((n + 255) / 256), 256, 0, stream>>>(
        c_fw_final, H_, Ccat, 2 * H_, 0, B_, H_);
    copy_cols_f32<<<dim3((n + 255) / 256), 256, 0, stream>>>(
        c_bw_final, H_, Ccat, 2 * H_, H_, B_, H_);
    wmma_gemm_nt<<<dim3(H_ / 32, B_ / 32), 32, 0, stream>>>(
        Ccat, 2 * H_, bWini, 2 * H_, cdec, H_, b_init, 2 * H_, H_);
    tanh_vec<<<dim3((n + 255) / 256), 256, 0, stream>>>(cdec, hdec, n);
  }

  // ---- precompute Wa_h = h_s @ w_a^T   [S*B, H]
  wmma_gemm_nt<<<dim3(H_ / 32, S_ * B_ / 32), 32, 0, stream>>>(
      hsb, 2 * H_, bWa, 2 * H_, WaH, H_, nullptr, 2 * H_, H_);

  // ---- decoder scan
  cur = 0;
  for (int t = 0; t < T_; ++t) {
    // vh = h @ v_a^T
    wmma_gemm_nt<<<dim3(H_ / 32, B_ / 32), 32, 0, stream>>>(
        hdec + cur * nz, H_, bVa, H_, vh, H_, nullptr, H_, H_);
    // attention -> ct written into Acat[:, H:3H]
    attn_kernel<<<B_, S_, 0, stream>>>(WaH, vh, u_a, s_mask, hsb,
                                       Acat + H_, 3 * H_);
    // LSTM cell; new h written both to hdec ping-pong and Acat[:, 0:H]
    lstm_step<<<lgrid, 32, 0, stream>>>(
        hdec + cur * nz, cdec + cur * nz, bWhhD, Xd + (size_t)t * B_ * 4 * H_,
        hdec + (1 - cur) * nz, cdec + (1 - cur) * nz, Acat, 3 * H_, 0, H_);
    // logits = [h|ct] @ [w_b|v_b]^T   (dominant GEMM, bf16 WMMA)
    wmma_gemm_nt<<<dim3((V_ + 31) / 32, B_ / 32), 32, 0, stream>>>(
        Acat, 3 * H_, bWcat, 3 * H_, lgts, V_, nullptr, 3 * H_, V_);
    softmax_rows<<<B_, 256, 0, stream>>>(lgts, p + (size_t)t * B_ * V_, V_);
    cur = 1 - cur;
  }

  loss_kernel<<<1, B_, 0, stream>>>(p, t_ind, t_mask, loss);
}
